// DualStreamMamba_83451214561943
// MI455X (gfx1250) — compile-verified, matrix operand order checked by reading, not run on hardware
//
#include <hip/hip_runtime.h>
#include <hip/hip_bf16.h>

#define N_CHUNKS 16
#define LCHUNK   1024
#define D_MODEL  512
#define D_INNER  1024
#define D_STATE  16
#define D_CONVK  4
#define DT_RANK  32
#define NHEAD    8
#define DHEAD    64
#define NTOK     (N_CHUNKS * LCHUNK)   /* 16384 rows */

typedef __attribute__((ext_vector_type(16))) __bf16 v16bf;
typedef __attribute__((ext_vector_type(8)))  float  v8f;

struct FragU { uint4 lo, hi; };

__device__ __forceinline__ unsigned short f2bf(float f) {
  unsigned u = __float_as_uint(f);
  u += 0x7FFFu + ((u >> 16) & 1u);           // round-to-nearest-even
  return (unsigned short)(u >> 16);
}

// Load one 16-bf16 WMMA fragment from an LDS row laid out K-contiguously.
// Per ISA 16-bit A-matrix 16x32 layout: elements 0..7 -> K = 8*half + e,
// elements 8..15 -> K = 16 + 8*half + (e-8). Two 16B ds_load_b128 per lane.
__device__ __forceinline__ v16bf load_frag(const unsigned short* p, int kh) {
  FragU f;
  f.lo = *(const uint4*)(p + 8 * kh);
  f.hi = *(const uint4*)(p + 16 + 8 * kh);
  return __builtin_bit_cast(v16bf, f);
}

// ---------------------------------------------------------------------------
// Generic WMMA GEMM: C[M,N] = A[M,K] * B[K,N] (+ bias) (+ beta*Cold)
// block = 256 threads = 8 wave32 in a 2(M)x4(N) grid; block tile 64x128;
// each wave owns a 32x32 output tile -> 4 x v_wmma_f32_16x16x32_bf16 per
// K-step with 2 A-frags and 2 B-frags (operand reuse). LDS is double
// buffered so the next tile's global loads overlap the current WMMAs and
// only one barrier per K-step is needed.
// ---------------------------------------------------------------------------
__global__ __launch_bounds__(256)
void gemm_bf16_wmma(const float* __restrict__ A, int lda,
                    const float* __restrict__ B, int ldb,
                    float* __restrict__ C, int ldc,
                    const float* __restrict__ bias,
                    int M, int N, int K, int beta)
{
  __shared__ __align__(16) unsigned short As[2][64][40];    // [m][k], pad 8
  __shared__ __align__(16) unsigned short Bs[2][128][40];   // [n][k] transposed

  const int t       = threadIdx.x;
  const int m0      = blockIdx.y * 64;
  const int n0      = blockIdx.x * 128;
  const int wave    = t >> 5;
  const int lane    = t & 31;
  const int lane_lo = lane & 15;
  const int kh      = lane >> 4;       // half-wave select (K sub-range)
  const int wn      = wave & 3;        // 4 waves across N (32 cols each)
  const int wm      = wave >> 2;       // 2 waves across M (32 rows each)

  v8f acc00 = {0.f,0.f,0.f,0.f,0.f,0.f,0.f,0.f};
  v8f acc01 = {0.f,0.f,0.f,0.f,0.f,0.f,0.f,0.f};
  v8f acc10 = {0.f,0.f,0.f,0.f,0.f,0.f,0.f,0.f};
  v8f acc11 = {0.f,0.f,0.f,0.f,0.f,0.f,0.f,0.f};

  // ---- staging helpers (fp32 global -> bf16 LDS) ----
  auto stageA = [&](int k0, int buf) {
#pragma unroll
    for (int it = 0; it < 2; ++it) {           // 64x32 = 2048 elems
      int e   = (t + it * 256) << 2;
      int row = e >> 5, col = e & 31;
      int gm  = m0 + row;
      float4 f = make_float4(0.f, 0.f, 0.f, 0.f);
      if (gm < M) f = *(const float4*)(A + (size_t)gm * lda + k0 + col);
      uint2 p;
      p.x = (unsigned)f2bf(f.x) | ((unsigned)f2bf(f.y) << 16);
      p.y = (unsigned)f2bf(f.z) | ((unsigned)f2bf(f.w) << 16);
      *(uint2*)&As[buf][row][col] = p;
    }
  };
  auto stageB = [&](int k0, int buf) {
#pragma unroll
    for (int it = 0; it < 4; ++it) {           // 32x128 = 4096 elems
      int e  = (t + it * 256) << 2;
      int kk = e >> 7, nn = e & 127;
      float4 f = make_float4(0.f, 0.f, 0.f, 0.f);
      if ((n0 + nn) < N)
        f = *(const float4*)(B + (size_t)(k0 + kk) * ldb + n0 + nn);
      Bs[buf][nn + 0][kk] = f2bf(f.x);
      Bs[buf][nn + 1][kk] = f2bf(f.y);
      Bs[buf][nn + 2][kk] = f2bf(f.z);
      Bs[buf][nn + 3][kk] = f2bf(f.w);
    }
  };

  // prologue: fill buffer 0
  stageA(0, 0);
  stageB(0, 0);
  __syncthreads();

  int buf = 0;
  for (int k0 = 0; k0 < K; k0 += 32) {
    int nxt = buf ^ 1;
    if (k0 + 32 < K) {                 // prefetch next K tile into other buffer
      stageA(k0 + 32, nxt);
      stageB(k0 + 32, nxt);
      // cache-prefetch the tile after that
      if (k0 + 64 < K) {
        int gm = m0 + (t & 63);
        if (gm < M) __builtin_prefetch(A + (size_t)gm * lda + k0 + 64, 0, 1);
      }
    }

    v16bf a0 = load_frag(&As[buf][wm * 32 + lane_lo][0],       kh);
    v16bf a1 = load_frag(&As[buf][wm * 32 + 16 + lane_lo][0],  kh);
    v16bf b0 = load_frag(&Bs[buf][wn * 32 + lane_lo][0],       kh);
    v16bf b1 = load_frag(&Bs[buf][wn * 32 + 16 + lane_lo][0],  kh);
    acc00 = __builtin_amdgcn_wmma_f32_16x16x32_bf16(false, a0, false, b0,
                                                    (short)0, acc00, false, false);
    acc01 = __builtin_amdgcn_wmma_f32_16x16x32_bf16(false, a0, false, b1,
                                                    (short)0, acc01, false, false);
    acc10 = __builtin_amdgcn_wmma_f32_16x16x32_bf16(false, a1, false, b0,
                                                    (short)0, acc10, false, false);
    acc11 = __builtin_amdgcn_wmma_f32_16x16x32_bf16(false, a1, false, b1,
                                                    (short)0, acc11, false, false);
    __syncthreads();
    buf = nxt;
  }

  // C/D layout: VGPR r -> row r + 8*half, col = lane_lo within a 16x16 tile
  const int colbase = n0 + wn * 32 + lane_lo;
  const int rowbase = m0 + wm * 32 + 8 * kh;
#pragma unroll
  for (int mi = 0; mi < 2; ++mi) {
#pragma unroll
    for (int ni = 0; ni < 2; ++ni) {
      const v8f& a = (mi == 0) ? (ni == 0 ? acc00 : acc01)
                               : (ni == 0 ? acc10 : acc11);
      int col = colbase + ni * 16;
#pragma unroll
      for (int r = 0; r < 8; ++r) {
        int row = rowbase + mi * 16 + r;
        if (row < M && col < N) {
          float v = a[r];
          if (bias) v += bias[col];
          if (beta) v += C[(size_t)row * ldc + col];
          C[(size_t)row * ldc + col] = v;
        }
      }
    }
  }
}

// ---------------------------------------------------------------------------
// Causal depthwise conv1d (k=4) + SiLU.  xz: [NTOK, 2*D_INNER] (xi = cols 0..1023)
// ---------------------------------------------------------------------------
__global__ __launch_bounds__(256)
void conv_silu_kernel(const float* __restrict__ xz,
                      const float* __restrict__ conv_w,
                      const float* __restrict__ conv_b,
                      float* __restrict__ xc)
{
  int idx = blockIdx.x * 256 + threadIdx.x;
  if (idx >= NTOK * D_INNER) return;
  int c   = idx & (D_INNER - 1);
  int row = idx >> 10;
  int l   = row & (LCHUNK - 1);
  const float* base = xz + (size_t)(row - l) * (2 * D_INNER) + c;
  float acc = conv_b[c];
#pragma unroll
  for (int kk = 0; kk < D_CONVK; ++kk) {
    int ll = l - (D_CONVK - 1) + kk;
    if (ll >= 0) acc += base[(size_t)ll * (2 * D_INNER)] * conv_w[c * D_CONVK + kk];
  }
  xc[idx] = acc / (1.0f + __expf(-acc));      // silu
}

// ---------------------------------------------------------------------------
// Chunk means over L:  chunks [16,1024,512] -> means [16,512]
// ---------------------------------------------------------------------------
__global__ __launch_bounds__(256)
void mean_kernel(const float* __restrict__ chunks, float* __restrict__ means)
{
  int idx = blockIdx.x * 256 + threadIdx.x;
  if (idx >= N_CHUNKS * D_MODEL) return;
  int d  = idx & (D_MODEL - 1);
  int ch = idx / D_MODEL;
  const float* p = chunks + (size_t)ch * LCHUNK * D_MODEL + d;
  float s = 0.f;
  for (int l = 0; l < LCHUNK; ++l) s += p[(size_t)l * D_MODEL];
  means[idx] = s * (1.0f / LCHUNK);
}

// ---------------------------------------------------------------------------
// Selective scan. 1 block per chunk, 256 threads x 4 channels; 16 states in
// registers per channel; B_t/C_t broadcast via LDS each timestep. Fuses
// softplus(dt), D-skip and SiLU(z) gate -> ybuf [NTOK, D_INNER].
// ---------------------------------------------------------------------------
__global__ __launch_bounds__(256)
void scan_kernel(const float* __restrict__ xc,  const float* __restrict__ dbc,
                 const float* __restrict__ dtr, const float* __restrict__ dt_b,
                 const float* __restrict__ A_log, const float* __restrict__ Dp,
                 const float* __restrict__ xz,  float* __restrict__ ybuf)
{
  const int chunk = blockIdx.x;
  const int t     = threadIdx.x;
  __shared__ float sB[D_STATE], sC[D_STATE];

  float h[4][D_STATE];
  float Arow[4][D_STATE];
  float dtb[4], dpv[4];
#pragma unroll
  for (int j = 0; j < 4; ++j) {
    int c = t * 4 + j;
    dtb[j] = dt_b[c];
    dpv[j] = Dp[c];
#pragma unroll
    for (int s = 0; s < D_STATE; ++s) {
      Arow[j][s] = -__expf(A_log[c * D_STATE + s]);
      h[j][s] = 0.f;
    }
  }

  for (int l = 0; l < LCHUNK; ++l) {
    const int row = chunk * LCHUNK + l;
    if (t < D_STATE)            sB[t]      = dbc[(size_t)row * 64 + DT_RANK + t];
    else if (t < 2 * D_STATE)   sC[t - 16] = dbc[(size_t)row * 64 + DT_RANK + D_STATE + (t - 16)];
    __syncthreads();
#pragma unroll
    for (int j = 0; j < 4; ++j) {
      int c = t * 4 + j;
      float dtv = dtr[(size_t)row * D_INNER + c] + dtb[j];
      dtv = (dtv > 20.f) ? dtv : __logf(1.f + __expf(dtv));   // softplus
      float xv = xc[(size_t)row * D_INNER + c];
      float dx = dtv * xv;
      float y = 0.f;
#pragma unroll
      for (int s = 0; s < D_STATE; ++s) {
        float dA = __expf(dtv * Arow[j][s]);
        float hn = dA * h[j][s] + dx * sB[s];
        h[j][s] = hn;
        y += hn * sC[s];
      }
      float z  = xz[(size_t)row * (2 * D_INNER) + D_INNER + c];
      float sz = z / (1.f + __expf(-z));
      ybuf[(size_t)row * D_INNER + c] = (y + xv * dpv[j]) * sz;
    }
    __syncthreads();
  }
}

// ---------------------------------------------------------------------------
// Masked cross attention over 16 chunk summaries. 1 thread per (chunk,l,head).
// ---------------------------------------------------------------------------
__global__ __launch_bounds__(256)
void attn_kernel(const float* __restrict__ q, const float* __restrict__ kb,
                 const float* __restrict__ vb, float* __restrict__ attnv)
{
  int idx = blockIdx.x * 256 + threadIdx.x;
  if (idx >= NTOK * NHEAD) return;
  int hh  = idx & (NHEAD - 1);
  int tok = idx >> 3;
  int i   = tok >> 10;                       // chunk index
  const float* qp = q + (size_t)tok * D_MODEL + hh * DHEAD;
  float qr[DHEAD];
#pragma unroll
  for (int d = 0; d < DHEAD; ++d) qr[d] = qp[d];

  float sc[N_CHUNKS];
  float mx = -3.4e38f;
#pragma unroll
  for (int j = 0; j < N_CHUNKS; ++j) {
    if (j == i) { sc[j] = -3.4e38f; continue; }
    const float* kp = kb + (size_t)j * D_MODEL + hh * DHEAD;
    float s = 0.f;
    for (int d = 0; d < DHEAD; ++d) s += qr[d] * kp[d];
    sc[j] = s * 0.125f;                      // 1/sqrt(64)
    mx = fmaxf(mx, sc[j]);
  }
  float den = 0.f;
#pragma unroll
  for (int j = 0; j < N_CHUNKS; ++j) {
    float e = (j == i) ? 0.f : __expf(sc[j] - mx);
    sc[j] = e; den += e;
  }
  float inv = 1.f / den;
  float* op = attnv + (size_t)tok * D_MODEL + hh * DHEAD;
  for (int d = 0; d < DHEAD; ++d) {
    float a = 0.f;
#pragma unroll
    for (int j = 0; j < N_CHUNKS; ++j)
      a += sc[j] * vb[(size_t)j * D_MODEL + hh * DHEAD + d];
    op[d] = a * inv;
  }
}

// ---------------------------------------------------------------------------
extern "C" void kernel_launch(void* const* d_in, const int* in_sizes, int n_in,
                              void* d_out, int out_size, void* d_ws, size_t ws_size,
                              hipStream_t stream)
{
  (void)in_sizes; (void)n_in; (void)out_size; (void)ws_size;
  const float* chunks   = (const float*)d_in[0];
  const float* in_proj  = (const float*)d_in[1];
  const float* conv_w   = (const float*)d_in[2];
  const float* conv_b   = (const float*)d_in[3];
  const float* x_proj   = (const float*)d_in[4];
  const float* dt_w     = (const float*)d_in[5];
  const float* dt_b     = (const float*)d_in[6];
  const float* A_log    = (const float*)d_in[7];
  const float* Dp       = (const float*)d_in[8];
  const float* out_proj = (const float*)d_in[9];
  const float* sum_w    = (const float*)d_in[10];
  const float* sum_b    = (const float*)d_in[11];
  const float* wq       = (const float*)d_in[12];
  const float* wk       = (const float*)d_in[13];
  const float* wv       = (const float*)d_in[14];
  const float* wo       = (const float*)d_in[15];
  float* out = (float*)d_out;

  float* ws = (float*)d_ws;
  size_t off = 0;
  float* xz    = ws + off; off += (size_t)NTOK * 2 * D_INNER;   // [16384,2048]
  float* xc    = ws + off; off += (size_t)NTOK * D_INNER;       // [16384,1024]
  float* dbc   = ws + off; off += (size_t)NTOK * 64;            // [16384,64]
  float* dtr   = ws + off; off += (size_t)NTOK * D_INNER;       // [16384,1024]
  float* ybuf  = ws + off; off += (size_t)NTOK * D_INNER;       // [16384,1024]
  float* qbuf  = ws + off; off += (size_t)NTOK * D_MODEL;       // [16384,512]
  float* av    = ws + off; off += (size_t)NTOK * D_MODEL;       // [16384,512]
  float* means = ws + off; off += (size_t)N_CHUNKS * D_MODEL;
  float* summ  = ws + off; off += (size_t)N_CHUNKS * D_MODEL;
  float* kbuf  = ws + off; off += (size_t)N_CHUNKS * D_MODEL;
  float* vbuf  = ws + off; off += (size_t)N_CHUNKS * D_MODEL;

  dim3 blk(256);
  auto gemm = [&](const float* A, int lda, const float* B, int ldb, float* C,
                  int ldc, const float* bias, int M, int N, int K, int beta) {
    dim3 grid((N + 127) / 128, (M + 63) / 64);
    gemm_bf16_wmma<<<grid, blk, 0, stream>>>(A, lda, B, ldb, C, ldc, bias, M, N, K, beta);
  };

  // 1. xz = chunks @ in_proj                      [16384,512]x[512,2048]
  gemm(chunks, D_MODEL, in_proj, 2 * D_INNER, xz, 2 * D_INNER, nullptr,
       NTOK, 2 * D_INNER, D_MODEL, 0);
  // 2. xc = silu(causal_conv(xi))
  conv_silu_kernel<<<(NTOK * D_INNER + 255) / 256, blk, 0, stream>>>(xz, conv_w, conv_b, xc);
  // 3. dbc = xc @ x_proj                          [16384,1024]x[1024,64]
  gemm(xc, D_INNER, x_proj, 64, dbc, 64, nullptr, NTOK, 64, D_INNER, 0);
  // 4. dt_raw = dbc[:, :32] @ dt_w                [16384,32]x[32,1024]
  gemm(dbc, 64, dt_w, D_INNER, dtr, D_INNER, nullptr, NTOK, D_INNER, DT_RANK, 0);
  // 5. selective scan -> ybuf (fused softplus / D-skip / silu(z) gate)
  scan_kernel<<<N_CHUNKS, blk, 0, stream>>>(xc, dbc, dtr, dt_b, A_log, Dp, xz, ybuf);
  // 6. tok = ybuf @ out_proj  -> d_out            [16384,1024]x[1024,512]
  gemm(ybuf, D_INNER, out_proj, D_MODEL, out, D_MODEL, nullptr, NTOK, D_MODEL, D_INNER, 0);
  // 7. means; 8. summaries = means @ sum_w + sum_b
  mean_kernel<<<(N_CHUNKS * D_MODEL + 255) / 256, blk, 0, stream>>>(chunks, means);
  gemm(means, D_MODEL, sum_w, D_MODEL, summ, D_MODEL, sum_b, N_CHUNKS, D_MODEL, D_MODEL, 0);
  // 9-11. q/k/v projections
  gemm(out,  D_MODEL, wq, D_MODEL, qbuf, D_MODEL, nullptr, NTOK,     D_MODEL, D_MODEL, 0);
  gemm(summ, D_MODEL, wk, D_MODEL, kbuf, D_MODEL, nullptr, N_CHUNKS, D_MODEL, D_MODEL, 0);
  gemm(summ, D_MODEL, wv, D_MODEL, vbuf, D_MODEL, nullptr, N_CHUNKS, D_MODEL, D_MODEL, 0);
  // 12. masked softmax cross-attention over 16 summaries
  attn_kernel<<<(NTOK * NHEAD + 255) / 256, blk, 0, stream>>>(qbuf, kbuf, vbuf, av);
  // 13. d_out += av @ wo
  gemm(av, D_MODEL, wo, D_MODEL, out, D_MODEL, nullptr, NTOK, D_MODEL, D_MODEL, 1);
}
